// NonLocalBlock_7352984011247
// MI455X (gfx1250) — matrix operand order repaired and checked by reading, not compiled, not run on hardware
//
#include <hip/hip_runtime.h>
#include <hip/hip_bf16.h>
#include <math.h>

// ---------------------------------------------------------------------------
// NonLocalBlock for MI455X (gfx1250, wave32, WMMA bf16, f32 accum).
// Workspace (256 MB): phi/theta/g bf16 [8][512][2048], S f32 [8][2048][2048],
// attn bf16 [8][2048][2048], yT bf16 [8][512][2048].
// GEMM: block 128x64x32, 8 waves (4Mx2N), wave tile 32x32 = 4 WMMA/Kstep.
// LDS: double-buffered; A [128][32] K-contiguous, B transposed [64][32]
// K-contiguous -> every WMMA fragment is 2x ds_load_b128 per lane.
// Pipeline: load tile t+1 (global->regs, or async DMA ->LDS), WMMA tile t,
// commit regs->LDS, one barrier per K-step.
// ---------------------------------------------------------------------------

typedef __attribute__((ext_vector_type(16))) __bf16 v16bf;
typedef __attribute__((ext_vector_type(8)))  __bf16 v8bf;
typedef __attribute__((ext_vector_type(8)))  float  v8f;

#define CH    512
#define INTER 256
#define NB    8
#define HW    4096
#define HW2   2048

#define BM 128
#define BN 64
#define BK 32
#define LDAK 40   // halves per A row (32 data + 8 pad) -> 80B, 16B aligned
#define LDBK 40
#define ASZ  (BM * LDAK)
#define BSZ  (BN * LDBK)

// --- WMMA fragment loaders (wave32 layouts, ISA 7.12.2) --------------------

__device__ __forceinline__ v16bf load_a_frag(const __bf16* As, int m0, int lane) {
  const int mm = lane & 15;
  const int kh = (lane >> 4) << 3;                 // 0 or 8
  const __bf16* row = As + (m0 + mm) * LDAK;
  union { v16bf v; v8bf h[2]; } u;
  u.h[0] = *(const v8bf*)(row + kh);
  u.h[1] = *(const v8bf*)(row + 16 + kh);
  return u.v;
}

__device__ __forceinline__ v16bf load_b_frag(const __bf16* Bt, int n0, int lane) {
  const int nn = lane & 15;
  const int kb = (lane >> 4) << 3;                 // 0 or 8
  const __bf16* p = Bt + (n0 + nn) * LDBK + kb;
  union { v16bf v; v8bf h[2]; } u;
  u.h[0] = *(const v8bf*)p;
  u.h[1] = *(const v8bf*)(p + 16);
  return u.v;
}

#define WMMA_BF16(acc, a, b) \
  __builtin_amdgcn_wmma_f32_16x16x32_bf16(false, (a), false, (b), (short)0, (acc), false, false)

// --- staged-tile register containers ---------------------------------------

struct SA { unsigned w[8]; };   // 128x32 bf16 tile, 8 packed u32 per thread
struct SB { unsigned w[4]; };   // 64x32  bf16 tile, 4 packed u32 per thread

__device__ __forceinline__ unsigned pack2bf(float a, float b) {
  union { __bf16 h[2]; unsigned u; } c;
  c.h[0] = (__bf16)a; c.h[1] = (__bf16)b;
  return c.u;
}

// A direct [M][K] f32 source (K-contiguous): r = i>>4, cp = (i&15)*2
__device__ __forceinline__ SA loadA_f32(const float* W, size_t ld, int m0g,
                                        int kk, int tid) {
  SA s;
#pragma unroll
  for (int j = 0; j < 8; ++j) {
    const int i = tid + j * 256;
    const int r = i >> 4, cp = (i & 15) << 1;
    const float* p = W + (size_t)(m0g + r) * ld + kk + cp;
    s.w[j] = pack2bf(p[0], p[1]);
  }
  return s;
}
__device__ __forceinline__ void storeA_direct(__bf16* As, int tid, const SA& s) {
#pragma unroll
  for (int j = 0; j < 8; ++j) {
    const int i = tid + j * 256;
    const int r = i >> 4, cp = (i & 15) << 1;
    *(unsigned*)&As[r * LDAK + cp] = s.w[j];
  }
}

// A transposed bf16 source [K][N] -> As[n][k]: n = i&127, kp = (i>>7)*2
__device__ __forceinline__ SA loadA_bf16T(const unsigned short* T, size_t ld,
                                          int kk, int n0g, int tid) {
  SA s;
#pragma unroll
  for (int j = 0; j < 8; ++j) {
    const int i = tid + j * 256;
    const int n = i & 127, kp = (i >> 7) << 1;
    const unsigned lo = T[(size_t)(kk + kp) * ld + n0g + n];
    const unsigned hi = T[(size_t)(kk + kp + 1) * ld + n0g + n];
    s.w[j] = lo | (hi << 16);
  }
  return s;
}
__device__ __forceinline__ void storeA_T(__bf16* As, int tid, const SA& s) {
#pragma unroll
  for (int j = 0; j < 8; ++j) {
    const int i = tid + j * 256;
    const int n = i & 127, kp = (i >> 7) << 1;
    *(unsigned*)&As[n * LDAK + kp] = s.w[j];
  }
}

// B transpose [K][N] -> Bt[n][k]: n = i&63, kp = (i>>6)*2
__device__ __forceinline__ SB loadB_bf16(const unsigned short* S, size_t ld,
                                         int kk, int n0g, int tid) {
  SB s;
#pragma unroll
  for (int j = 0; j < 4; ++j) {
    const int i = tid + j * 256;
    const int n = i & 63, kp = (i >> 6) << 1;
    const unsigned lo = S[(size_t)(kk + kp) * ld + n0g + n];
    const unsigned hi = S[(size_t)(kk + kp + 1) * ld + n0g + n];
    s.w[j] = lo | (hi << 16);
  }
  return s;
}
__device__ __forceinline__ SB loadB_f32(const float* S, size_t ld,
                                        int kk, int n0g, int tid) {
  SB s;
#pragma unroll
  for (int j = 0; j < 4; ++j) {
    const int i = tid + j * 256;
    const int n = i & 63, kp = (i >> 6) << 1;
    const float lo = S[(size_t)(kk + kp) * ld + n0g + n];
    const float hi = S[(size_t)(kk + kp + 1) * ld + n0g + n];
    s.w[j] = pack2bf(lo, hi);
  }
  return s;
}
__device__ __forceinline__ void storeB(__bf16* Bt, int tid, const SB& s) {
#pragma unroll
  for (int j = 0; j < 4; ++j) {
    const int i = tid + j * 256;
    const int n = i & 63, kp = (i >> 6) << 1;
    *(unsigned*)&Bt[n * LDBK + kp] = s.w[j];
  }
}

// --- CDNA5 async copy (global -> LDS, ASYNCcnt protocol) -------------------
__device__ __forceinline__ void async_copy_b128(const void* gptr, void* lptr) {
  asm volatile("global_load_async_to_lds_b128 %0, %1, off"
               :: "v"((unsigned)(size_t)lptr),
                  "v"((unsigned long long)(size_t)gptr)
               : "memory");
}
__device__ __forceinline__ void wait_async0() {
  asm volatile("s_wait_asynccnt 0x0" ::: "memory");
}

// --- shared GEMM core ------------------------------------------------------
#define GEMM_PROLOG()                                                     \
  const int tid = threadIdx.x, lane = tid & 31, wave = tid >> 5;          \
  const int m0 = (wave >> 1) * 32, n0 = (wave & 1) * 32;                  \
  v8f acc00 = {}; v8f acc01 = {}; v8f acc10 = {}; v8f acc11 = {}

#define GEMM_MMA(Asb, Btb)                                                \
  do {                                                                    \
    const v16bf a0 = load_a_frag((Asb), m0, lane);                        \
    const v16bf a1 = load_a_frag((Asb), m0 + 16, lane);                   \
    const v16bf b0 = load_b_frag((Btb), n0, lane);                        \
    const v16bf b1 = load_b_frag((Btb), n0 + 16, lane);                   \
    acc00 = WMMA_BF16(acc00, a0, b0);                                     \
    acc01 = WMMA_BF16(acc01, a0, b1);                                     \
    acc10 = WMMA_BF16(acc10, a1, b0);                                     \
    acc11 = WMMA_BF16(acc11, a1, b1);                                     \
  } while (0)

// --- Kernel 1: three 1x1 convs  Out[o,p] = sum_c W[o,c]*X[c,p] -------------
// M=256, N=4096, K=512.  grid = (64, 2, 24)
__global__ __launch_bounds__(256)
void k_conv3(const float* __restrict__ q, const float* __restrict__ k,
             const float* __restrict__ v,
             const float* __restrict__ Wphi, const float* __restrict__ Wtheta,
             const float* __restrict__ Wg,
             __bf16* __restrict__ phi, __bf16* __restrict__ theta,
             __bf16* __restrict__ g) {
  __shared__ __attribute__((aligned(16))) __bf16 As[2 * ASZ];
  __shared__ __attribute__((aligned(16))) __bf16 Bt[2 * BSZ];
  const int z = blockIdx.z;
  const int which = z % 3, b = z / 3;
  const float* W = (which == 0) ? Wphi : (which == 1) ? Wtheta : Wg;
  const float* X = ((which == 0) ? q : (which == 1) ? k : v) + (size_t)b * CH * HW;
  __bf16* Out = ((which == 0) ? phi : (which == 1) ? theta : g) + (size_t)b * INTER * HW;

  const int o0 = blockIdx.y * BM;
  const int p0 = blockIdx.x * BN;
  GEMM_PROLOG();

  SA sa = loadA_f32(W, CH, o0, 0, tid);
  SB sb = loadB_f32(X, HW, 0, p0, tid);
  storeA_direct(As, tid, sa);
  storeB(Bt, tid, sb);
  __syncthreads();

  const int nk = CH / BK;
  for (int t = 0; t < nk; ++t) {
    const int cur = t & 1, nxt = cur ^ 1;
    if (t + 1 < nk) {
      sa = loadA_f32(W, CH, o0, (t + 1) * BK, tid);
      sb = loadB_f32(X, HW, (t + 1) * BK, p0, tid);
    }
    GEMM_MMA(As + cur * ASZ, Bt + cur * BSZ);
    if (t + 1 < nk) {
      storeA_direct(As + nxt * ASZ, tid, sa);
      storeB(Bt + nxt * BSZ, tid, sb);
    }
    __syncthreads();
  }

  const int rb = (lane >> 4) << 3;
  const int col = lane & 15;
#pragma unroll
  for (int r = 0; r < 8; ++r) {
    const size_t r0 = (size_t)(o0 + m0 + rb + r) * HW + p0 + n0;
    const size_t r1 = (size_t)(o0 + m0 + 16 + rb + r) * HW + p0 + n0;
    Out[r0 + col]      = (__bf16)acc00[r];
    Out[r0 + 16 + col] = (__bf16)acc01[r];
    Out[r1 + col]      = (__bf16)acc10[r];
    Out[r1 + 16 + col] = (__bf16)acc11[r];
  }
}

// --- Kernel 2: S[n,m] = sum_c theta[c,n]*phi[c,m] --------------------------
// M=2048(n), N=2048(m), K=512.  grid = (32, 16, 8)
__global__ __launch_bounds__(256)
void k_scores(const __bf16* __restrict__ theta, const __bf16* __restrict__ phi,
              float* __restrict__ S) {
  __shared__ __attribute__((aligned(16))) __bf16 As[2 * ASZ];
  __shared__ __attribute__((aligned(16))) __bf16 Bt[2 * BSZ];
  const int b = blockIdx.z;
  const unsigned short* Tb = (const unsigned short*)(theta + (size_t)b * CH * HW2);
  const unsigned short* Pb = (const unsigned short*)(phi + (size_t)b * CH * HW2);
  float* Sb = S + (size_t)b * HW2 * HW2;

  const int nrow0 = blockIdx.y * BM;
  const int mcol0 = blockIdx.x * BN;
  GEMM_PROLOG();

  SA sa = loadA_bf16T(Tb, HW2, 0, nrow0, tid);
  SB sb = loadB_bf16(Pb, HW2, 0, mcol0, tid);
  storeA_T(As, tid, sa);
  storeB(Bt, tid, sb);
  __syncthreads();

  const int nk = CH / BK;
  for (int t = 0; t < nk; ++t) {
    const int cur = t & 1, nxt = cur ^ 1;
    if (t + 1 < nk) {
      sa = loadA_bf16T(Tb, HW2, (t + 1) * BK, nrow0, tid);
      sb = loadB_bf16(Pb, HW2, (t + 1) * BK, mcol0, tid);
    }
    GEMM_MMA(As + cur * ASZ, Bt + cur * BSZ);
    if (t + 1 < nk) {
      storeA_T(As + nxt * ASZ, tid, sa);
      storeB(Bt + nxt * BSZ, tid, sb);
    }
    __syncthreads();
  }

  const int rb = (lane >> 4) << 3;
  const int col = lane & 15;
#pragma unroll
  for (int r = 0; r < 8; ++r) {
    const size_t r0 = (size_t)(nrow0 + m0 + rb + r) * HW2 + mcol0 + n0;
    const size_t r1 = (size_t)(nrow0 + m0 + 16 + rb + r) * HW2 + mcol0 + n0;
    Sb[r0 + col]      = acc00[r];
    Sb[r0 + 16 + col] = acc01[r];
    Sb[r1 + col]      = acc10[r];
    Sb[r1 + 16 + col] = acc11[r];
  }
}

// --- Kernel 3: softmax over n (dim=1), per column m ------------------------
__global__ __launch_bounds__(256)
void k_softmax(const float* __restrict__ S, __bf16* __restrict__ attn) {
  const int b = blockIdx.y;
  const int m0 = blockIdx.x * 64;
  const int tx = threadIdx.x & 63, ty = threadIdx.x >> 6;
  const float* Sb = S + (size_t)b * HW2 * HW2;
  __bf16* Ab = attn + (size_t)b * HW2 * HW2;
  const int m = m0 + tx;

  __shared__ float red[4][64];

  float mx = -INFINITY;
  for (int n = ty; n < HW2; n += 4) mx = fmaxf(mx, Sb[(size_t)n * HW2 + m]);
  red[ty][tx] = mx;
  __syncthreads();
  const float m4 = fmaxf(fmaxf(red[0][tx], red[1][tx]),
                         fmaxf(red[2][tx], red[3][tx]));
  __syncthreads();

  float s = 0.f;
  for (int n = ty; n < HW2; n += 4) s += __expf(Sb[(size_t)n * HW2 + m] - m4);
  red[ty][tx] = s;
  __syncthreads();
  const float inv = 1.0f / (red[0][tx] + red[1][tx] + red[2][tx] + red[3][tx]);

  for (int n = ty; n < HW2; n += 4)
    Ab[(size_t)n * HW2 + m] = (__bf16)(__expf(Sb[(size_t)n * HW2 + m] - m4) * inv);
}

// --- Kernel 4: yT[c,m] = sum_n g[c,n]*attn[n,m] ----------------------------
// M=512(c), N=2048(m), K=2048.  grid = (32, 4, 8)
__global__ __launch_bounds__(256)
void k_ygemm(const __bf16* __restrict__ g, const __bf16* __restrict__ attn,
             __bf16* __restrict__ yT) {
  __shared__ __attribute__((aligned(16))) __bf16 As[2 * ASZ];
  __shared__ __attribute__((aligned(16))) __bf16 Bt[2 * BSZ];
  const int b = blockIdx.z;
  const __bf16* Gb = g + (size_t)b * CH * HW2;
  const unsigned short* Ab = (const unsigned short*)(attn + (size_t)b * HW2 * HW2);
  __bf16* Yb = yT + (size_t)b * CH * HW2;

  const int c0 = blockIdx.y * BM;
  const int mc0 = blockIdx.x * BN;
  GEMM_PROLOG();

  // A tile via CDNA5 async DMA (b128 per lane), B tile via reg staging.
#pragma unroll
  for (int j = 0; j < 2; ++j) {
    const int i = tid + j * 256;
    const int r = i >> 2, co = (i & 3) << 3;
    async_copy_b128(Gb + (size_t)(c0 + r) * HW2 + co, &As[r * LDAK + co]);
  }
  SB sb = loadB_bf16(Ab, HW2, 0, mc0, tid);
  storeB(Bt, tid, sb);
  wait_async0();
  __syncthreads();

  const int nk = HW2 / BK;
  for (int t = 0; t < nk; ++t) {
    const int cur = t & 1, nxt = cur ^ 1;
    if (t + 1 < nk) {
      const int kk = (t + 1) * BK;
#pragma unroll
      for (int j = 0; j < 2; ++j) {        // async DMA next A tile -> LDS[nxt]
        const int i = tid + j * 256;
        const int r = i >> 2, co = (i & 3) << 3;
        async_copy_b128(Gb + (size_t)(c0 + r) * HW2 + kk + co,
                        &As[nxt * ASZ + r * LDAK + co]);
      }
      sb = loadB_bf16(Ab, HW2, kk, mc0, tid);
    }
    GEMM_MMA(As + cur * ASZ, Bt + cur * BSZ);
    if (t + 1 < nk) {
      storeB(Bt + nxt * BSZ, tid, sb);
      wait_async0();
    }
    __syncthreads();
  }

  const int rb = (lane >> 4) << 3;
  const int col = lane & 15;
#pragma unroll
  for (int r = 0; r < 8; ++r) {
    const size_t r0 = (size_t)(c0 + m0 + rb + r) * HW2 + mc0 + n0;
    const size_t r1 = (size_t)(c0 + m0 + 16 + rb + r) * HW2 + mc0 + n0;
    Yb[r0 + col]      = (__bf16)acc00[r];
    Yb[r0 + 16 + col] = (__bf16)acc01[r];
    Yb[r1 + col]      = (__bf16)acc10[r];
    Yb[r1 + 16 + col] = (__bf16)acc11[r];
  }
}

// --- Kernel 5: out = W_mask @ y2 + q + v -----------------------------------
// y2 = yT viewed [INTER][HW].  M=512, N=4096, K=256.  grid = (64, 4, 8)
__global__ __launch_bounds__(256)
void k_mask(const __bf16* __restrict__ yT, const float* __restrict__ Wm,
            const float* __restrict__ q, const float* __restrict__ v,
            float* __restrict__ out) {
  __shared__ __attribute__((aligned(16))) __bf16 As[2 * ASZ];
  __shared__ __attribute__((aligned(16))) __bf16 Bt[2 * BSZ];
  const int b = blockIdx.z;
  const unsigned short* Yb = (const unsigned short*)(yT + (size_t)b * CH * HW2);

  const int co0 = blockIdx.y * BM;
  const int p0 = blockIdx.x * BN;
  GEMM_PROLOG();

  SA sa = loadA_f32(Wm, INTER, co0, 0, tid);
  SB sb = loadB_bf16(Yb, HW, 0, p0, tid);
  storeA_direct(As, tid, sa);
  storeB(Bt, tid, sb);
  __syncthreads();

  const int nk = INTER / BK;
  for (int t = 0; t < nk; ++t) {
    const int cur = t & 1, nxt = cur ^ 1;
    if (t + 1 < nk) {
      sa = loadA_f32(Wm, INTER, co0, (t + 1) * BK, tid);
      sb = loadB_bf16(Yb, HW, (t + 1) * BK, p0, tid);
    }
    GEMM_MMA(As + cur * ASZ, Bt + cur * BSZ);
    if (t + 1 < nk) {
      storeA_direct(As + nxt * ASZ, tid, sa);
      storeB(Bt + nxt * BSZ, tid, sb);
    }
    __syncthreads();
  }

  const int rb = (lane >> 4) << 3;
  const int col = lane & 15;
#pragma unroll
  for (int r = 0; r < 8; ++r) {
    const size_t base = (size_t)b * CH * HW;
    const size_t r0 = base + (size_t)(co0 + m0 + rb + r) * HW + p0 + n0;
    const size_t r1 = base + (size_t)(co0 + m0 + 16 + rb + r) * HW + p0 + n0;
    out[r0 + col]      = acc00[r] + q[r0 + col] + v[r0 + col];
    out[r0 + 16 + col] = acc01[r] + q[r0 + 16 + col] + v[r0 + 16 + col];
    out[r1 + col]      = acc10[r] + q[r1 + col] + v[r1 + col];
    out[r1 + 16 + col] = acc11[r] + q[r1 + 16 + col] + v[r1 + 16 + col];
  }
}

// ---------------------------------------------------------------------------

extern "C" void kernel_launch(void* const* d_in, const int* in_sizes, int n_in,
                              void* d_out, int out_size, void* d_ws, size_t ws_size,
                              hipStream_t stream) {
  (void)in_sizes; (void)n_in; (void)out_size; (void)ws_size;
  const float* q      = (const float*)d_in[0];
  const float* k      = (const float*)d_in[1];
  const float* v      = (const float*)d_in[2];
  const float* Wphi   = (const float*)d_in[3];
  const float* Wtheta = (const float*)d_in[4];
  const float* Wg     = (const float*)d_in[5];
  const float* Wmask  = (const float*)d_in[6];
  float* out = (float*)d_out;

  char* w = (char*)d_ws;
  const size_t qkv_b = (size_t)NB * CH * HW2 * sizeof(__bf16);   // 16.78 MB
  __bf16* phi   = (__bf16*)w;  w += qkv_b;
  __bf16* theta = (__bf16*)w;  w += qkv_b;
  __bf16* g     = (__bf16*)w;  w += qkv_b;
  float*  S     = (float*)w;   w += (size_t)NB * HW2 * HW2 * sizeof(float);
  __bf16* attn  = (__bf16*)w;  w += (size_t)NB * HW2 * HW2 * sizeof(__bf16);
  __bf16* yT    = (__bf16*)w;

  const dim3 blk(256);
  k_conv3  <<<dim3(HW / BN, INTER / BM, NB * 3), blk, 0, stream>>>(
      q, k, v, Wphi, Wtheta, Wg, phi, theta, g);
  k_scores <<<dim3(HW2 / BN, HW2 / BM, NB), blk, 0, stream>>>(theta, phi, S);
  k_softmax<<<dim3(HW2 / 64, NB), blk, 0, stream>>>(S, attn);
  k_ygemm  <<<dim3(HW2 / BN, CH / BM, NB), blk, 0, stream>>>(g, attn, yT);
  k_mask   <<<dim3(HW / BN, CH / BM, NB), blk, 0, stream>>>(yT, Wmask, q, v, out);
}
